// Pka_acidic_view_56899726738020
// MI455X (gfx1250) — matrix operand order repaired and checked by reading, not compiled, not run on hardware
//
#include <hip/hip_runtime.h>
#include <hip/hip_bf16.h>

typedef __bf16 bf16;
typedef bf16  v16bf __attribute__((ext_vector_type(16)));
typedef bf16  v8bf  __attribute__((ext_vector_type(8)));
typedef float v8f   __attribute__((ext_vector_type(8)));

// ---------------- problem constants ----------------
#define Vn   100000
#define En   400000
#define Gf   200
#define Nf   74
#define NEf  12
#define Ln   2
#define NGn  4000
#define KP1  96     // padded K for 74 / 86
#define KP2  224    // padded K for 200
#define NP2  256    // padded N for 200
#define NP6  640    // padded N for 600
#define BFS  224    // bf16 activation row stride

static inline int ceil_div(int a, int b) { return (a + b - 1) / b; }

// ---------------- elementwise / utility kernels ----------------
__global__ void fill_kernel(float* p, int n, float v) {
  int i = blockIdx.x * blockDim.x + threadIdx.x;
  if (i < n) p[i] = v;
}

__global__ void mask_kernel(const float* __restrict__ nfeat, float* __restrict__ mask, int V_, int N_) {
  int v = blockIdx.x * blockDim.x + threadIdx.x;
  if (v >= V_) return;
  const float* r = nfeat + (size_t)v * N_;
  float s = r[N_ - 4] + r[N_ - 3] + r[N_ - 2] + r[N_ - 1];
  float m = s * (1.0f - r[0]);
  mask[v] = 1.0f / m - 1.0f;
}

// pack fp32 weight (K x Nc, row-major) into zero-padded bf16 (Kp x Np)
__global__ void pack_w_kernel(const float* __restrict__ W, int K, int Nc,
                              bf16* __restrict__ out, int Kp, int Np) {
  int i = blockIdx.x * blockDim.x + threadIdx.x;
  if (i >= Kp * Np) return;
  int r = i / Np, c = i % Np;
  float v = (r < K && c < Nc) ? W[(size_t)r * Nc + c] : 0.0f;
  out[i] = (bf16)v;
}

// node feats -> bf16 V x 96 (zero padded)
__global__ void pack_node_kernel(const float* __restrict__ nfeat, bf16* __restrict__ out, int V_) {
  int i = blockIdx.x * blockDim.x + threadIdx.x;
  if (i >= V_ * KP1) return;
  int v = i / KP1, c = i % KP1;
  float x = (c < Nf) ? nfeat[(size_t)v * Nf + c] : 0.0f;
  out[i] = (bf16)x;
}

// [node_feats[src] | edge_feats] -> bf16 E x 96 (zero padded)
__global__ void pack_he1A_kernel(const float* __restrict__ nfeat, const float* __restrict__ efeat,
                                 const int* __restrict__ srcI, bf16* __restrict__ out, int E_) {
  int i = blockIdx.x * blockDim.x + threadIdx.x;
  if (i >= E_ * KP1) return;
  int e = i / KP1, c = i % KP1;
  float x;
  if (c < Nf)            x = nfeat[(size_t)srcI[e] * Nf + c];
  else if (c < Nf + NEf) x = efeat[(size_t)e * NEf + (c - Nf)];
  else                   x = 0.0f;
  out[i] = (bf16)x;
}

// x = elu(c) -> bf16 (stride BFS; pads assumed pre-zeroed)
__global__ void elu_pack_kernel(const float* __restrict__ c, bf16* __restrict__ xbf, int V_, int G_) {
  int i = blockIdx.x * blockDim.x + threadIdx.x;
  if (i >= V_ * G_) return;
  int v = i / G_, g = i % G_;
  float x = c[i];
  x = x > 0.0f ? x : (__expf(x) - 1.0f);
  xbf[(size_t)v * BFS + g] = (bf16)x;
}

// ---------------- WMMA bf16 GEMM ----------------
// Block = 128 threads (4 waves) computes a 64x64 output tile.
// Wave w owns 16-col strip w; loops over 4 16-row subtiles -> 4 wmma per K-step.
// A staged row-major (64x32), B staged transposed (64 cols x 32 k) so every
// per-lane fragment is contiguous 16B chunks -> ds_load_b128.
// Double-buffered LDS: stage K-chunk kc+32 while computing chunk kc.
// mode 0: plain (+bias); mode 1: leaky-relu(+bias);
// mode 2: attention scatter: acc[dst[m]*Nc+n] += attn[m]*(v+bias)
__global__ __launch_bounds__(128)
void wmma_gemm_kernel(const bf16* __restrict__ A, int lda,
                      const bf16* __restrict__ B, int ldb,
                      const float* __restrict__ bias,
                      float* __restrict__ C, int ldc,
                      bf16* __restrict__ Cbf, int ldcbf,
                      int M, int Kp, int Nc, int mode,
                      const int* __restrict__ edst,
                      const float* __restrict__ attn,
                      float* __restrict__ acc) {
  __shared__ bf16 sA[2][64 * 32];   // row r (0..63), k (0..31): sA[p][r*32+k]
  __shared__ bf16 sBt[2][64 * 32];  // col n (0..63), k (0..31): sBt[p][n*32+k]
  const int tid  = threadIdx.x;
  const int wave = tid >> 5;
  const int lane = tid & 31;
  const int m = lane & 15;       // A row / B col / C col within tile
  const int h = lane >> 4;
  const int row0 = blockIdx.x << 6;
  const int col0 = blockIdx.y << 6;

  v8f acc0 = {0.f,0.f,0.f,0.f,0.f,0.f,0.f,0.f};
  v8f acc1 = acc0, acc2 = acc0, acc3 = acc0;

  // stage one 64x32 A tile + transposed 32x64 B tile for K-chunk kc
  auto stage = [&](bf16* dA, bf16* dBt, int kc) {
    // A: 64 rows x 32 cols bf16 = 1024 dwords (16 dwords/row), 8 per thread
#pragma unroll
    for (int t = 0; t < 8; ++t) {
      int i = tid + t * 128;
      int r = i >> 4, c2 = (i & 15) << 1;
      int gr = row0 + r; if (gr >= M) gr = M - 1;
      ((unsigned int*)dA)[i] = *(const unsigned int*)&A[(size_t)gr * lda + kc + c2];
    }
    // B: 32 rows x 64 cols bf16 = 1024 dwords (32 dwords/row), 8 per thread
#pragma unroll
    for (int t = 0; t < 8; ++t) {
      int i = tid + t * 128;
      int r = i >> 5, c2 = (i & 31) << 1;
      unsigned int d = *(const unsigned int*)&B[(size_t)(kc + r) * ldb + col0 + c2];
      bf16 lo = ((const bf16*)&d)[0];
      bf16 hi = ((const bf16*)&d)[1];
      dBt[c2 * 32 + r] = lo;
      dBt[(c2 + 1) * 32 + r] = hi;
    }
  };

  int p = 0;
  stage(sA[0], sBt[0], 0);
  __syncthreads();

  for (int kc = 0; kc < Kp; kc += 32) {
    if (kc + 32 < Kp) stage(sA[p ^ 1], sBt[p ^ 1], kc + 32);

    // B fragment: k = 8h..8h+7 and 16+8h..16+8h+7 at sBt[p][(wave*16+m)*32 + 8h]
    const bf16* bp = &sBt[p][((wave << 4) + m) * 32 + (h << 3)];
    v8bf b0 = *(const v8bf*)bp;
    v8bf b1 = *(const v8bf*)(bp + 16);
    v16bf bfrag = __builtin_shufflevector(b0, b1, 0,1,2,3,4,5,6,7,8,9,10,11,12,13,14,15);

#pragma unroll
    for (int rt = 0; rt < 4; ++rt) {
      const bf16* ap = &sA[p][((rt << 4) + m) * 32 + (h << 3)];
      v8bf a0 = *(const v8bf*)ap;                 // k = 8h .. 8h+7
      v8bf a1 = *(const v8bf*)(ap + 16);          // k = 16+8h .. 16+8h+7
      v16bf afrag = __builtin_shufflevector(a0, a1, 0,1,2,3,4,5,6,7,8,9,10,11,12,13,14,15);
      v8f* pacc = (rt == 0) ? &acc0 : (rt == 1) ? &acc1 : (rt == 2) ? &acc2 : &acc3;
      *pacc = __builtin_amdgcn_wmma_f32_16x16x32_bf16(false, afrag, false, bfrag,
                                                      (short)0, *pacc, false, false);
    }
    __syncthreads();   // releases sA[p]/sBt[p] for overwrite, publishes sA[p^1]/sBt[p^1]
    p ^= 1;
  }

  const int gn = col0 + (wave << 4) + m;
  if (gn < Nc) {
#pragma unroll
    for (int rt = 0; rt < 4; ++rt) {
      v8f av = (rt == 0) ? acc0 : (rt == 1) ? acc1 : (rt == 2) ? acc2 : acc3;
#pragma unroll
      for (int r = 0; r < 8; ++r) {
        int gm = row0 + (rt << 4) + r + (h << 3);
        if (gm >= M) continue;
        float v = av[r];
        if (bias) v += bias[gn];
        if (mode == 1) v = v > 0.0f ? v : 0.01f * v;
        if (mode == 2) {
          atomicAdd(&acc[(size_t)edst[gm] * Nc + gn], attn[gm] * v);
        } else {
          if (C)   C[(size_t)gm * ldc + gn] = v;
          if (Cbf) Cbf[(size_t)gm * ldcbf + gn] = (bf16)v;
        }
      }
    }
  }
}

// ---------------- edge logits (wave per edge) ----------------
__global__ __launch_bounds__(256)
void ctx_logits_kernel(const float* __restrict__ hv, const bf16* __restrict__ he1bf,
                       const float* __restrict__ W, const float* __restrict__ b,
                       const int* __restrict__ dstI, float* __restrict__ logit, int E_) {
  int wave = threadIdx.x >> 5, lane = threadIdx.x & 31;
  int e = blockIdx.x * 8 + wave;
  if (e >= E_) return;
  int d = dstI[e];
  float s = 0.0f;
  for (int g = lane; g < Gf; g += 32)
    s += hv[(size_t)d * Gf + g] * W[g] + (float)he1bf[(size_t)e * BFS + g] * W[Gf + g];
  for (int off = 16; off; off >>= 1) s += __shfl_xor(s, off, 32);
  if (lane == 0) {
    float v = s + b[0];
    logit[e] = v > 0.0f ? v : 0.01f * v;
  }
}

__global__ __launch_bounds__(256)
void gnn_logits_kernel(const float* __restrict__ nf, const float* __restrict__ W,
                       const float* __restrict__ b, const int* __restrict__ srcI,
                       const int* __restrict__ dstI, float* __restrict__ logit, int E_) {
  int wave = threadIdx.x >> 5, lane = threadIdx.x & 31;
  int e = blockIdx.x * 8 + wave;
  if (e >= E_) return;
  int d = dstI[e], s0 = srcI[e];
  float s = 0.0f;
  for (int g = lane; g < Gf; g += 32)
    s += nf[(size_t)d * Gf + g] * W[g] + nf[(size_t)s0 * Gf + g] * W[Gf + g];
  for (int off = 16; off; off >>= 1) s += __shfl_xor(s, off, 32);
  if (lane == 0) {
    float v = s + b[0];
    logit[e] = v > 0.0f ? v : 0.01f * v;
  }
}

// ---------------- edge softmax passes ----------------
__device__ inline void atomicMaxF(float* addr, float value) {
  if (value >= 0.0f) atomicMax((int*)addr, __float_as_int(value));
  else               atomicMin((unsigned int*)addr, __float_as_uint(value));
}

__global__ void segmax_kernel(const float* __restrict__ logit, const int* __restrict__ dstI,
                              float* __restrict__ segmax, int E_) {
  int e = blockIdx.x * blockDim.x + threadIdx.x;
  if (e < E_) atomicMaxF(&segmax[dstI[e]], logit[e]);
}
__global__ void segsum_kernel(const float* __restrict__ logit, const int* __restrict__ dstI,
                              const float* __restrict__ segmax, float* __restrict__ segsum, int E_) {
  int e = blockIdx.x * blockDim.x + threadIdx.x;
  if (e >= E_) return;
  int d = dstI[e];
  atomicAdd(&segsum[d], __expf(logit[e] - segmax[d]));
}
__global__ void attn_kernel(const float* __restrict__ logit, const int* __restrict__ dstI,
                            const float* __restrict__ segmax, const float* __restrict__ segsum,
                            float* __restrict__ attn, int E_) {
  int e = blockIdx.x * blockDim.x + threadIdx.x;
  if (e >= E_) return;
  int d = dstI[e];
  attn[e] = __expf(logit[e] - segmax[d]) / segsum[d];
}

// c[dst] += attn * hvp[src]
__global__ void scatter_attn_kernel(const float* __restrict__ hvp, const float* __restrict__ attn,
                                    const int* __restrict__ srcI, const int* __restrict__ dstI,
                                    float* __restrict__ c, int E_) {
  int i = blockIdx.x * blockDim.x + threadIdx.x;
  if (i >= E_ * Gf) return;
  int e = i / Gf, g = i % Gf;
  atomicAdd(&c[(size_t)dstI[e] * Gf + g], attn[e] * hvp[(size_t)srcI[e] * Gf + g]);
}

// ---------------- GRU gate fusion ----------------
__global__ void gru_combine_kernel(const float* __restrict__ gi, const float* __restrict__ gh,
                                   const float* __restrict__ bih, const float* __restrict__ bhh,
                                   const float* __restrict__ h, float* __restrict__ nf,
                                   bf16* __restrict__ nfbf, int V_) {
  int i = blockIdx.x * blockDim.x + threadIdx.x;
  if (i >= V_ * Gf) return;
  int v = i / Gf, g = i % Gf;
  const float* giv = gi + (size_t)v * 3 * Gf;
  const float* ghv = gh + (size_t)v * 3 * Gf;
  float ir = giv[g]          + bih[g],          hr = ghv[g]          + bhh[g];
  float iz = giv[Gf + g]     + bih[Gf + g],     hz = ghv[Gf + g]     + bhh[Gf + g];
  float in = giv[2 * Gf + g] + bih[2 * Gf + g], hn = ghv[2 * Gf + g] + bhh[2 * Gf + g];
  float r = 1.0f / (1.0f + __expf(-(ir + hr)));
  float z = 1.0f / (1.0f + __expf(-(iz + hz)));
  float n = tanhf(in + r * hn);
  float out = (1.0f - z) * n + z * h[i];
  out = out > 0.0f ? out : 0.0f;
  nf[i] = out;
  nfbf[(size_t)v * BFS + g] = (bf16)out;
}

// ---------------- readout ----------------
__global__ __launch_bounds__(256)
void readout_kernel(const float* __restrict__ nf, const float* __restrict__ predW,
                    const float* __restrict__ predb, const float* __restrict__ mask,
                    const int* __restrict__ gids, float* __restrict__ gsum,
                    float* __restrict__ outAtom, int V_) {
  int wave = threadIdx.x >> 5, lane = threadIdx.x & 31;
  int v = blockIdx.x * 8 + wave;
  if (v >= V_) return;
  float s = 0.0f;
  for (int g = lane; g < Gf; g += 32) s += nf[(size_t)v * Gf + g] * predW[g];
  for (int off = 16; off; off >>= 1) s += __shfl_xor(s, off, 32);
  if (lane == 0) {
    float raw = s + predb[0];
    float apk = -(raw + mask[v]);
    outAtom[v] = -apk;
    atomicAdd(&gsum[gids[v]], __expf(apk * 2.3025850929940457f));
  }
}

__global__ void finalize_kernel(const float* __restrict__ gsum, float* __restrict__ out, int NG_) {
  int g = blockIdx.x * blockDim.x + threadIdx.x;
  if (g < NG_) out[g] = -log10f(gsum[g]);
}

// ---------------- host orchestration ----------------
extern "C" void kernel_launch(void* const* d_in, const int* in_sizes, int n_in,
                              void* d_out, int out_size, void* d_ws, size_t ws_size,
                              hipStream_t stream) {
  const float* node_feats = (const float*)d_in[0];
  const float* edge_feats = (const float*)d_in[1];
  const float* pn_W  = (const float*)d_in[2];  const float* pn_b  = (const float*)d_in[3];
  const float* pe1_W = (const float*)d_in[4];  const float* pe1_b = (const float*)d_in[5];
  const float* pe2_W = (const float*)d_in[6];  const float* pe2_b = (const float*)d_in[7];
  const float* et_W  = (const float*)d_in[8];  const float* et_b  = (const float*)d_in[9];
  const float* g0Wih = (const float*)d_in[10]; const float* g0Whh = (const float*)d_in[11];
  const float* g0bih = (const float*)d_in[12]; const float* g0bhh = (const float*)d_in[13];
  const float* gpeW  = (const float*)d_in[14]; const float* gpeB  = (const float*)d_in[15];
  const float* gpnW  = (const float*)d_in[16]; const float* gpnB  = (const float*)d_in[17];
  const float* gWih  = (const float*)d_in[18]; const float* gWhh  = (const float*)d_in[19];
  const float* gbih  = (const float*)d_in[20]; const float* gbhh  = (const float*)d_in[21];
  const float* predW = (const float*)d_in[22]; const float* predB = (const float*)d_in[23];
  const int* srcI = (const int*)d_in[24];
  const int* dstI = (const int*)d_in[25];
  const int* gids = (const int*)d_in[26];
  float* out = (float*)d_out;
  float* outG = out;            // [0, NGn)
  float* outA = out + NGn;      // [NGn, NGn+Vn)

  // -------- workspace bump allocator --------
  char* ws = (char*)d_ws;
  size_t off = 0;
  auto alloc = [&](size_t bytes) -> char* {
    char* p = ws + off;
    off = (off + bytes + 255) & ~(size_t)255;
    return p;
  };
  float* mask   = (float*)alloc((size_t)Vn * 4);
  float* hv     = (float*)alloc((size_t)Vn * Gf * 4);
  bf16*  hv_bf  = (bf16*) alloc((size_t)Vn * BFS * 2);       // aliased as nf_bf after gru0
  bf16*  Ahe1   = (bf16*) alloc((size_t)En * KP1 * 2);       // aliased as x_bf later
  bf16*  x_bf   = Ahe1;
  bf16*  he1bf  = (bf16*) alloc((size_t)En * BFS * 2);
  float* logit  = (float*)alloc((size_t)En * 4);
  float* attn   = (float*)alloc((size_t)En * 4);
  float* segmax = (float*)alloc((size_t)Vn * 4);
  float* segsum = (float*)alloc((size_t)Vn * 4);
  float* cbuf   = (float*)alloc((size_t)Vn * Gf * 4);
  float* gi     = (float*)alloc((size_t)Vn * 3 * Gf * 4);
  float* gh     = (float*)alloc((size_t)Vn * 3 * Gf * 4);
  float* nf     = (float*)alloc((size_t)Vn * Gf * 4);
  float* hvp    = (float*)alloc((size_t)Vn * Gf * 4);
  float* gsum   = (float*)alloc((size_t)NGn * 4);
  bf16*  nodebf = (bf16*) alloc((size_t)Vn * KP1 * 2);
  bf16*  wpn  = (bf16*)alloc((size_t)KP1 * NP2 * 2);
  bf16*  wpe1 = (bf16*)alloc((size_t)KP1 * NP2 * 2);
  bf16*  wet  = (bf16*)alloc((size_t)KP2 * NP2 * 2);
  bf16*  wih0 = (bf16*)alloc((size_t)KP2 * NP6 * 2);
  bf16*  whh0 = (bf16*)alloc((size_t)KP2 * NP6 * 2);
  bf16*  wpnl[Ln]; bf16* wihl[Ln]; bf16* whhl[Ln];
  for (int l = 0; l < Ln; ++l) {
    wpnl[l] = (bf16*)alloc((size_t)KP2 * NP2 * 2);
    wihl[l] = (bf16*)alloc((size_t)KP2 * NP6 * 2);
    whhl[l] = (bf16*)alloc((size_t)KP2 * NP6 * 2);
  }
  (void)ws_size; (void)in_sizes; (void)n_in; (void)out_size;

  const int T = 256;
  auto gemm = [&](const bf16* A, int lda, const bf16* B, int Np, const float* bias,
                  float* C, int ldc, bf16* Cbf, int ldcbf, int M, int Kp, int Nc,
                  int mode, const int* edst, const float* at, float* acc) {
    dim3 grid(ceil_div(M, 64), Np / 64);
    wmma_gemm_kernel<<<grid, dim3(128), 0, stream>>>(A, lda, B, Np, bias, C, ldc, Cbf, ldcbf,
                                                     M, Kp, Nc, mode, edst, at, acc);
  };
  auto edge_softmax = [&](const float* lg) {
    fill_kernel<<<ceil_div(Vn, T), T, 0, stream>>>(segmax, Vn, -3.0e38f);
    hipMemsetAsync(segsum, 0, (size_t)Vn * 4, stream);
    segmax_kernel<<<ceil_div(En, T), T, 0, stream>>>(lg, dstI, segmax, En);
    segsum_kernel<<<ceil_div(En, T), T, 0, stream>>>(lg, dstI, segmax, segsum, En);
    attn_kernel  <<<ceil_div(En, T), T, 0, stream>>>(lg, dstI, segmax, segsum, attn, En);
  };
  auto gru = [&](const bf16* xb, const bf16* hb, const bf16* Wih, const bf16* Whh,
                 const float* bih, const float* bhh, const float* h) {
    gemm(xb, BFS, Wih, NP6, nullptr, gi, 3 * Gf, nullptr, 0, Vn, KP2, 3 * Gf, 0, nullptr, nullptr, nullptr);
    gemm(hb, BFS, Whh, NP6, nullptr, gh, 3 * Gf, nullptr, 0, Vn, KP2, 3 * Gf, 0, nullptr, nullptr, nullptr);
    gru_combine_kernel<<<ceil_div(Vn * Gf, T), T, 0, stream>>>(gi, gh, bih, bhh, h, nf, hv_bf, Vn);
  };

  // -------- zero bf16 pads / prep --------
  hipMemsetAsync(hv_bf, 0, (size_t)Vn * BFS * 2, stream);
  hipMemsetAsync(he1bf, 0, (size_t)En * BFS * 2, stream);

  mask_kernel<<<ceil_div(Vn, T), T, 0, stream>>>(node_feats, mask, Vn, Nf);

  // -------- pack weights --------
  pack_w_kernel<<<ceil_div(KP1 * NP2, T), T, 0, stream>>>(pn_W,  Nf,        Gf,     wpn,  KP1, NP2);
  pack_w_kernel<<<ceil_div(KP1 * NP2, T), T, 0, stream>>>(pe1_W, Nf + NEf,  Gf,     wpe1, KP1, NP2);
  pack_w_kernel<<<ceil_div(KP2 * NP2, T), T, 0, stream>>>(et_W,  Gf,        Gf,     wet,  KP2, NP2);
  pack_w_kernel<<<ceil_div(KP2 * NP6, T), T, 0, stream>>>(g0Wih, Gf,        3 * Gf, wih0, KP2, NP6);
  pack_w_kernel<<<ceil_div(KP2 * NP6, T), T, 0, stream>>>(g0Whh, Gf,        3 * Gf, whh0, KP2, NP6);
  for (int l = 0; l < Ln; ++l) {
    pack_w_kernel<<<ceil_div(KP2 * NP2, T), T, 0, stream>>>(gpnW + (size_t)l * Gf * Gf,     Gf, Gf,     wpnl[l], KP2, NP2);
    pack_w_kernel<<<ceil_div(KP2 * NP6, T), T, 0, stream>>>(gWih + (size_t)l * Gf * 3 * Gf, Gf, 3 * Gf, wihl[l], KP2, NP6);
    pack_w_kernel<<<ceil_div(KP2 * NP6, T), T, 0, stream>>>(gWhh + (size_t)l * Gf * 3 * Gf, Gf, 3 * Gf, whhl[l], KP2, NP6);
  }

  // -------- GetContext --------
  pack_node_kernel<<<ceil_div(Vn * KP1, T), T, 0, stream>>>(node_feats, nodebf, Vn);
  pack_he1A_kernel<<<ceil_div(En * KP1, T), T, 0, stream>>>(node_feats, edge_feats, srcI, Ahe1, En);

  gemm(nodebf, KP1, wpn,  NP2, pn_b,  hv, Gf, hv_bf, BFS, Vn, KP1, Gf, 1, nullptr, nullptr, nullptr);
  gemm(Ahe1,   KP1, wpe1, NP2, pe1_b, nullptr, 0, he1bf, BFS, En, KP1, Gf, 1, nullptr, nullptr, nullptr);

  ctx_logits_kernel<<<ceil_div(En, 8), 256, 0, stream>>>(hv, he1bf, pe2_W, pe2_b, dstI, logit, En);
  edge_softmax(logit);

  hipMemsetAsync(cbuf, 0, (size_t)Vn * Gf * 4, stream);
  // e = a * (he1 @ et_W + et_b), fused scatter: c[dst] += e
  gemm(he1bf, BFS, wet, NP2, et_b, nullptr, 0, nullptr, 0, En, KP2, Gf, 2, dstI, attn, cbuf);

  hipMemsetAsync(x_bf, 0, (size_t)Vn * BFS * 2, stream);  // pads stay zero for all layers
  elu_pack_kernel<<<ceil_div(Vn * Gf, T), T, 0, stream>>>(cbuf, x_bf, Vn, Gf);
  gru(x_bf, hv_bf, wih0, whh0, g0bih, g0bhh, hv);          // writes nf + nf_bf(=hv_bf)

  // -------- GNN layers --------
  for (int l = 0; l < Ln; ++l) {
    gnn_logits_kernel<<<ceil_div(En, 8), 256, 0, stream>>>(nf, gpeW + (size_t)l * 2 * Gf,
                                                           gpeB + l, srcI, dstI, logit, En);
    edge_softmax(logit);
    gemm(hv_bf, BFS, wpnl[l], NP2, gpnB + (size_t)l * Gf, hvp, Gf, nullptr, 0,
         Vn, KP2, Gf, 0, nullptr, nullptr, nullptr);
    hipMemsetAsync(cbuf, 0, (size_t)Vn * Gf * 4, stream);
    scatter_attn_kernel<<<ceil_div(En * Gf, T), T, 0, stream>>>(hvp, attn, srcI, dstI, cbuf, En);
    elu_pack_kernel<<<ceil_div(Vn * Gf, T), T, 0, stream>>>(cbuf, x_bf, Vn, Gf);
    gru(x_bf, hv_bf, wihl[l], whhl[l], gbih + (size_t)l * 3 * Gf, gbhh + (size_t)l * 3 * Gf, nf);
  }

  // -------- readout --------
  hipMemsetAsync(gsum, 0, (size_t)NGn * 4, stream);
  readout_kernel<<<ceil_div(Vn, 8), 256, 0, stream>>>(nf, predW, predB, mask, gids, gsum, outA, Vn);
  finalize_kernel<<<ceil_div(NGn, T), T, 0, stream>>>(gsum, outG, NGn);
}
